// Decoder_64716567216602
// MI455X (gfx1250) — matrix operand order repaired and checked by reading, not compiled, not run on hardware
//
#include <hip/hip_runtime.h>
#include <cstddef>
#include <cstdint>

// ---------------------------------------------------------------------------
// Types for CDNA5 WMMA (gfx1250, wave32)
// ---------------------------------------------------------------------------
typedef __attribute__((ext_vector_type(16))) __bf16 v16bf;
typedef __attribute__((ext_vector_type(8)))  float  v8f;

union FragAB {
    unsigned short u[16];
    uint4          q[2];
    v16bf          v;
};

// ---------------------------------------------------------------------------
// CDNA5 async global->LDS copy (ASYNCcnt path), with sync fallback
// builtin signature: void(v4i AS1*, v4i AS3*, imm int offset, imm int cpol)
// ---------------------------------------------------------------------------
#if defined(__has_builtin)
#if __has_builtin(__builtin_amdgcn_global_load_async_to_lds_b128)
#define USE_ASYNC_LDS 1
#endif
#if __has_builtin(__builtin_amdgcn_s_wait_asynccnt)
#define HAVE_WAIT_ASYNC_BUILTIN 1
#endif
#endif

typedef int v4i_ __attribute__((vector_size(16)));
typedef __attribute__((address_space(1))) v4i_ as1_v4i;
typedef __attribute__((address_space(3))) v4i_ as3_v4i;

__device__ __forceinline__ void async_cp16(const unsigned short* g, unsigned short* l) {
#ifdef USE_ASYNC_LDS
    // flat LDS address low 32 bits == LDS offset (ISA 10.2 aperture mapping)
    as1_v4i* gp = (as1_v4i*)(unsigned long long)g;
    as3_v4i* lp = (as3_v4i*)(unsigned int)(unsigned long long)l;
    __builtin_amdgcn_global_load_async_to_lds_b128(gp, lp, 0, 0);
#else
    *(uint4*)l = *(const uint4*)g;
#endif
}

__device__ __forceinline__ void wait_async0() {
#ifdef USE_ASYNC_LDS
#ifdef HAVE_WAIT_ASYNC_BUILTIN
    __builtin_amdgcn_s_wait_asynccnt(0);
#else
    asm volatile("s_wait_asynccnt 0x0" ::: "memory");
#endif
#endif
}

__device__ __forceinline__ unsigned short f2bf(float f) {
    unsigned int u = __float_as_uint(f);
    unsigned int r = (u + 0x7FFFu + ((u >> 16) & 1u)) >> 16;  // RNE
    return (unsigned short)r;
}

// ---------------------------------------------------------------------------
// Elementwise f32 -> bf16 (pairs packed as u32)
// ---------------------------------------------------------------------------
__global__ void cvt_f32_bf16_kernel(const float* __restrict__ in,
                                    unsigned short* __restrict__ out, int n2) {
    int i = blockIdx.x * blockDim.x + threadIdx.x;
    if (i < n2) {
        float2 f = ((const float2*)in)[i];
        unsigned int packed = (unsigned int)f2bf(f.x) | ((unsigned int)f2bf(f.y) << 16);
        ((unsigned int*)out)[i] = packed;
    }
}

// ---------------------------------------------------------------------------
// GEMM: out = act(A[MxK](bf16) @ W[KxN](bf16) + bias) (+ res)
// BM=128 BN=64 BK=32; 8 waves as 4x2, each owns a 32x32 tile (4 WMMAs/k-step).
// Double-buffered LDS fed by async global->LDS copies.
// ---------------------------------------------------------------------------
#define LDSA_STRIDE 40   // 32 + 8 pad (80B rows, 16B aligned)
#define LDSB_STRIDE 72   // 64 + 8 pad (144B rows, 16B aligned)

__global__ __launch_bounds__(256)
void gemm_bf16_kernel(const unsigned short* __restrict__ A,
                      const unsigned short* __restrict__ Wt,
                      const float* __restrict__ bias,
                      const float* __restrict__ res,
                      float* __restrict__ outF,
                      unsigned short* __restrict__ outB,
                      int M, int N, int K, int do_gelu) {
    __shared__ unsigned short lsA[2][128 * LDSA_STRIDE];
    __shared__ unsigned short lsB[2][32 * LDSB_STRIDE];

    const int tid  = threadIdx.x;
    const int lane = tid & 31;
    const int wave = tid >> 5;
    const int wm   = wave >> 1;   // 4 wave-rows of 32
    const int wn   = wave & 1;    // 2 wave-cols of 32
    const int bm   = blockIdx.y * 128;
    const int bn   = blockIdx.x * 64;
    const int hh   = lane >> 4;
    const int ml   = lane & 15;

    // per-thread streaming addresses (advance by constant strides)
    const int rA = tid >> 2;              // 0..63 (thread also covers row rA+64)
    const int cA = (tid & 3) * 8;
    const int rB = tid >> 3;              // 0..31
    const int cB = (tid & 7) * 8;
    const unsigned short* gA0 = A + (size_t)(bm + rA) * K + cA;
    const unsigned short* gA1 = A + (size_t)(bm + rA + 64) * K + cA;
    const unsigned short* gB  = Wt + (size_t)rB * N + bn + cB;
    const int lA0 = rA * LDSA_STRIDE + cA;
    const int lA1 = (rA + 64) * LDSA_STRIDE + cA;
    const int lB  = rB * LDSB_STRIDE + cB;
    const size_t strideB = (size_t)32 * N;

    v8f cacc[2][2] = {};
    const int nK = K >> 5;

    // prologue: stage 0
    async_cp16(gA0, &lsA[0][lA0]);
    async_cp16(gA1, &lsA[0][lA1]);
    async_cp16(gB,  &lsB[0][lB]);
    gA0 += 32; gA1 += 32; gB += strideB;
    wait_async0();
    __syncthreads();

    for (int kt = 0; kt < nK; ++kt) {
        const int cur = kt & 1;
        if (kt + 1 < nK) {
            const int nxt = cur ^ 1;
            async_cp16(gA0, &lsA[nxt][lA0]);
            async_cp16(gA1, &lsA[nxt][lA1]);
            async_cp16(gB,  &lsB[nxt][lB]);
            gA0 += 32; gA1 += 32; gB += strideB;
        }

        FragAB a0, a1, b0, b1;
        {
            const uint4* pa0 = (const uint4*)&lsA[cur][(wm * 32 + ml) * LDSA_STRIDE];
            a0.q[0] = pa0[hh]; a0.q[1] = pa0[hh + 2];
            const uint4* pa1 = (const uint4*)&lsA[cur][(wm * 32 + 16 + ml) * LDSA_STRIDE];
            a1.q[0] = pa1[hh]; a1.q[1] = pa1[hh + 2];
            #pragma unroll
            for (int e = 0; e < 16; ++e) {
                const int kk = (e < 8) ? (8 * hh + e) : (16 + 8 * hh + (e - 8));
                b0.u[e] = lsB[cur][kk * LDSB_STRIDE + wn * 32 + ml];
                b1.u[e] = lsB[cur][kk * LDSB_STRIDE + wn * 32 + 16 + ml];
            }
        }
        cacc[0][0] = __builtin_amdgcn_wmma_f32_16x16x32_bf16(
            false, a0.v, false, b0.v, (short)0, cacc[0][0], false, false);
        cacc[0][1] = __builtin_amdgcn_wmma_f32_16x16x32_bf16(
            false, a0.v, false, b1.v, (short)0, cacc[0][1], false, false);
        cacc[1][0] = __builtin_amdgcn_wmma_f32_16x16x32_bf16(
            false, a1.v, false, b0.v, (short)0, cacc[1][0], false, false);
        cacc[1][1] = __builtin_amdgcn_wmma_f32_16x16x32_bf16(
            false, a1.v, false, b1.v, (short)0, cacc[1][1], false, false);

        wait_async0();
        __syncthreads();
    }

    // epilogue
    #pragma unroll
    for (int i = 0; i < 2; ++i) {
        #pragma unroll
        for (int t = 0; t < 2; ++t) {
            const int ncol = bn + wn * 32 + t * 16 + ml;
            const float bv = bias ? bias[ncol] : 0.0f;
            #pragma unroll
            for (int r = 0; r < 8; ++r) {
                const int row = bm + wm * 32 + i * 16 + 8 * hh + r;
                float v = cacc[i][t][r] + bv;
                if (do_gelu)
                    v = 0.5f * v * (1.0f + erff(v * 0.70710678118654752f));
                size_t off = (size_t)row * N + ncol;
                if (res)  v += res[off];
                if (outF) outF[off] = v;
                if (outB) outB[off] = f2bf(v);
            }
        }
    }
}

// ---------------------------------------------------------------------------
// Flash attention: grid (Nseq/128, B*H); 8 waves x 16 q-rows/block.
// Online softmax over 32-key tiles; double-buffered async K/V staging.
// ---------------------------------------------------------------------------
__global__ __launch_bounds__(256)
void flash_attn_kernel(const unsigned short* __restrict__ Q,
                       const unsigned short* __restrict__ Km,
                       const unsigned short* __restrict__ Vm,
                       unsigned short* __restrict__ O,
                       int Nseq, int H, int causal, float scale) {
    const int Dh = 64;
    const int D  = H * Dh;   // 512
    __shared__ unsigned short lsK[2][32 * LDSB_STRIDE];
    __shared__ unsigned short lsV[2][32 * LDSB_STRIDE];
    __shared__ unsigned short lsP[8][16 * LDSA_STRIDE];

    const int tid  = threadIdx.x;
    const int lane = tid & 31;
    const int wave = tid >> 5;
    const int hh   = lane >> 4;
    const int ml   = lane & 15;

    const int bh = blockIdx.y;
    const int b  = bh / H;
    const int h  = bh % H;
    const int q0 = blockIdx.x * 128 + wave * 16;
    const size_t rowBase = (size_t)b * Nseq * D;
    const int hcol = h * Dh;

    // Q fragments: two K=32 chunks along Dh
    FragAB qf0, qf1;
    {
        const uint4* pq = (const uint4*)&Q[rowBase + (size_t)(q0 + ml) * D + hcol];
        qf0.q[0] = pq[hh];     qf0.q[1] = pq[hh + 2];
        qf1.q[0] = pq[4 + hh]; qf1.q[1] = pq[4 + hh + 2];
    }

    float m_run[8], l_run[8];
    v8f   oacc[4] = {};
    #pragma unroll
    for (int r = 0; r < 8; ++r) { m_run[r] = -3.0e38f; l_run[r] = 0.0f; }

    const int kend = causal ? (blockIdx.x * 128 + 128) : Nseq;

    // per-thread K/V streaming addresses
    const int rT = tid >> 3;
    const int cT = (tid & 7) * 8;
    const unsigned short* gK = Km + rowBase + (size_t)rT * D + hcol + cT;
    const unsigned short* gV = Vm + rowBase + (size_t)rT * D + hcol + cT;
    const int lT = rT * LDSB_STRIDE + cT;
    const size_t strideKV = (size_t)32 * D;

    // prologue: stage 0
    async_cp16(gK, &lsK[0][lT]);
    async_cp16(gV, &lsV[0][lT]);
    gK += strideKV; gV += strideKV;
    wait_async0();
    __syncthreads();

    for (int kb = 0; kb < kend; kb += 32) {
        const int cur = (kb >> 5) & 1;
        if (kb + 32 < kend) {
            const int nxt = cur ^ 1;
            async_cp16(gK, &lsK[nxt][lT]);
            async_cp16(gV, &lsV[nxt][lT]);
            gK += strideKV; gV += strideKV;
        }

        if (!causal || kb <= q0 + 15) {
            // S = Q K^T : B-frag lane = key, elements = d (contraction)
            v8f s0 = {}, s1 = {};
            #pragma unroll
            for (int c = 0; c < 2; ++c) {
                FragAB kt0, kt1;
                {
                    const uint4* pk0 = (const uint4*)&lsK[cur][ml * LDSB_STRIDE + c * 32];
                    const uint4* pk1 = (const uint4*)&lsK[cur][(16 + ml) * LDSB_STRIDE + c * 32];
                    kt0.q[0] = pk0[hh]; kt0.q[1] = pk0[hh + 2];
                    kt1.q[0] = pk1[hh]; kt1.q[1] = pk1[hh + 2];
                }
                const FragAB& qa = c ? qf1 : qf0;
                s0 = __builtin_amdgcn_wmma_f32_16x16x32_bf16(
                    false, qa.v, false, kt0.v, (short)0, s0, false, false);
                s1 = __builtin_amdgcn_wmma_f32_16x16x32_bf16(
                    false, qa.v, false, kt1.v, (short)0, s1, false, false);
            }

            // scale + causal mask + online softmax (rows span 16-lane halves)
            float p0[8], p1[8];
            #pragma unroll
            for (int r = 0; r < 8; ++r) {
                const int qrow = q0 + 8 * hh + r;
                float v0 = s0[r] * scale;
                float v1 = s1[r] * scale;
                if (causal) {
                    if (kb + ml > qrow)      v0 = -3.0e38f;
                    if (kb + 16 + ml > qrow) v1 = -3.0e38f;
                }
                float rm = fmaxf(v0, v1);
                rm = fmaxf(rm, __shfl_xor(rm, 1, 32));
                rm = fmaxf(rm, __shfl_xor(rm, 2, 32));
                rm = fmaxf(rm, __shfl_xor(rm, 4, 32));
                rm = fmaxf(rm, __shfl_xor(rm, 8, 32));
                const float mn   = fmaxf(m_run[r], rm);
                const float corr = __expf(m_run[r] - mn);
                m_run[r] = mn;
                l_run[r] *= corr;
                #pragma unroll
                for (int dt = 0; dt < 4; ++dt) oacc[dt][r] *= corr;
                const float e0 = __expf(v0 - mn);
                const float e1 = __expf(v1 - mn);
                p0[r] = e0; p1[r] = e1;
                float rs = e0 + e1;
                rs += __shfl_xor(rs, 1, 32);
                rs += __shfl_xor(rs, 2, 32);
                rs += __shfl_xor(rs, 4, 32);
                rs += __shfl_xor(rs, 8, 32);
                l_run[r] += rs;
            }

            // P: C-layout -> A-layout via wave-private LDS (DS in-order per wave)
            #pragma unroll
            for (int r = 0; r < 8; ++r) {
                lsP[wave][(8 * hh + r) * LDSA_STRIDE + ml]      = f2bf(p0[r]);
                lsP[wave][(8 * hh + r) * LDSA_STRIDE + 16 + ml] = f2bf(p1[r]);
            }
            FragAB pf;
            {
                const uint4* pp = (const uint4*)&lsP[wave][ml * LDSA_STRIDE];
                pf.q[0] = pp[hh];
                pf.q[1] = pp[hh + 2];
            }

            // O += P V : V B-frag lane = d, elements = key (contraction)
            #pragma unroll
            for (int dt = 0; dt < 4; ++dt) {
                FragAB vf;
                #pragma unroll
                for (int e = 0; e < 16; ++e) {
                    const int kk = (e < 8) ? (8 * hh + e) : (16 + 8 * hh + (e - 8));
                    vf.u[e] = lsV[cur][kk * LDSB_STRIDE + dt * 16 + ml];
                }
                oacc[dt] = __builtin_amdgcn_wmma_f32_16x16x32_bf16(
                    false, pf.v, false, vf.v, (short)0, oacc[dt], false, false);
            }
        }
        wait_async0();
        __syncthreads();
    }

    // finalize: O /= rowsum, store bf16
    #pragma unroll
    for (int dt = 0; dt < 4; ++dt) {
        #pragma unroll
        for (int r = 0; r < 8; ++r) {
            const int row = q0 + 8 * hh + r;
            const int col = hcol + dt * 16 + ml;
            O[rowBase + (size_t)row * D + col] = f2bf(oacc[dt][r] / l_run[r]);
        }
    }
}

// ---------------------------------------------------------------------------
// LayerNorm: one 512-wide row per 256-thread block; writes f32 + bf16
// ---------------------------------------------------------------------------
__global__ __launch_bounds__(256)
void layernorm_kernel(const float* __restrict__ in,
                      const float* __restrict__ g,
                      const float* __restrict__ bta,
                      float* __restrict__ outF,
                      unsigned short* __restrict__ outB, int D) {
    __shared__ float red[2][8];
    const int row = blockIdx.x;
    const int tid = threadIdx.x;
    const float* x = in + (size_t)row * D;
    const float v0 = x[tid];
    const float v1 = x[tid + 256];
    float s  = v0 + v1;
    float ss = v0 * v0 + v1 * v1;
    #pragma unroll
    for (int m = 1; m < 32; m <<= 1) {
        s  += __shfl_xor(s,  m, 32);
        ss += __shfl_xor(ss, m, 32);
    }
    const int wave = tid >> 5, lane = tid & 31;
    if (lane == 0) { red[0][wave] = s; red[1][wave] = ss; }
    __syncthreads();
    if (wave == 0) {
        s  = red[0][lane & 7];
        ss = red[1][lane & 7];
        #pragma unroll
        for (int m = 1; m < 8; m <<= 1) {
            s  += __shfl_xor(s,  m, 32);
            ss += __shfl_xor(ss, m, 32);
        }
        if (lane == 0) { red[0][0] = s; red[1][0] = ss; }
    }
    __syncthreads();
    s = red[0][0]; ss = red[1][0];
    const float mu   = s / (float)D;
    const float var  = ss / (float)D - mu * mu;
    const float rstd = rsqrtf(var + 1e-5f);
    const float y0 = (v0 - mu) * rstd * g[tid] + bta[tid];
    const float y1 = (v1 - mu) * rstd * g[tid + 256] + bta[tid + 256];
    const size_t off = (size_t)row * D;
    if (outF) { outF[off + tid] = y0; outF[off + tid + 256] = y1; }
    if (outB) { outB[off + tid] = f2bf(y0); outB[off + tid + 256] = f2bf(y1); }
}

// ---------------------------------------------------------------------------
// Orchestration
// ---------------------------------------------------------------------------
extern "C" void kernel_launch(void* const* d_in, const int* in_sizes, int n_in,
                              void* d_out, int out_size, void* d_ws, size_t ws_size,
                              hipStream_t stream) {
    (void)in_sizes; (void)n_in; (void)out_size; (void)ws_size;
    const int Bb = 2, Nseq = 2048, D = 512, H = 8, F = 2048;
    const int M = Bb * Nseq;              // 4096
    const float scale = 0.125f;           // 1/sqrt(64)

    const float* x    = (const float*)d_in[0];
    const float* enc  = (const float*)d_in[1];
    const float* m_wq = (const float*)d_in[2];  const float* m_bq = (const float*)d_in[3];
    const float* m_wk = (const float*)d_in[4];  const float* m_bk = (const float*)d_in[5];
    const float* m_wv = (const float*)d_in[6];  const float* m_bv = (const float*)d_in[7];
    const float* m_wo = (const float*)d_in[8];  const float* m_bo = (const float*)d_in[9];
    const float* c_wq = (const float*)d_in[10]; const float* c_bq = (const float*)d_in[11];
    const float* c_wk = (const float*)d_in[12]; const float* c_bk = (const float*)d_in[13];
    const float* c_wv = (const float*)d_in[14]; const float* c_bv = (const float*)d_in[15];
    const float* c_wo = (const float*)d_in[16]; const float* c_bo = (const float*)d_in[17];
    const float* ln1g = (const float*)d_in[18]; const float* ln1b = (const float*)d_in[19];
    const float* ln2g = (const float*)d_in[20]; const float* ln2b = (const float*)d_in[21];
    const float* ln3g = (const float*)d_in[22]; const float* ln3b = (const float*)d_in[23];
    const float* ffw1 = (const float*)d_in[24]; const float* ffb1 = (const float*)d_in[25];
    const float* ffw2 = (const float*)d_in[26]; const float* ffb2 = (const float*)d_in[27];

    // workspace carve (256B-aligned)
    char* wp = (char*)d_ws;
    auto alloc = [&](size_t bytes) -> void* {
        void* p = (void*)wp;
        wp += (bytes + 255) & ~(size_t)255;
        return p;
    };
    const float* wsrc[10] = { m_wq, m_wk, m_wv, m_wo, c_wq, c_wk, c_wv, c_wo, ffw1, ffw2 };
    size_t wcnt[10] = { (size_t)D*D, (size_t)D*D, (size_t)D*D, (size_t)D*D,
                        (size_t)D*D, (size_t)D*D, (size_t)D*D, (size_t)D*D,
                        (size_t)D*F, (size_t)F*D };
    unsigned short* wb[10];
    for (int i = 0; i < 10; ++i) wb[i] = (unsigned short*)alloc(wcnt[i] * 2);
    unsigned short* curb  = (unsigned short*)alloc((size_t)M * D * 2);
    unsigned short* encb  = (unsigned short*)alloc((size_t)M * D * 2);
    unsigned short* qb    = (unsigned short*)alloc((size_t)M * D * 2);
    unsigned short* kb    = (unsigned short*)alloc((size_t)M * D * 2);
    unsigned short* vb    = (unsigned short*)alloc((size_t)M * D * 2);
    unsigned short* attnb = (unsigned short*)alloc((size_t)M * D * 2);
    unsigned short* hbb   = (unsigned short*)alloc((size_t)M * F * 2);
    float*          sumf  = (float*)alloc((size_t)M * D * 4);
    float*          curf  = (float*)alloc((size_t)M * D * 4);

    auto cvt = [&](const float* src, unsigned short* dst, size_t n) {
        int n2 = (int)(n / 2);
        cvt_f32_bf16_kernel<<<(n2 + 255) / 256, 256, 0, stream>>>(src, dst, n2);
    };
    auto gemm = [&](const unsigned short* A, const unsigned short* W, const float* bias,
                    const float* res, float* oF, unsigned short* oB,
                    int Mx, int Nx, int Kx, int gelu) {
        dim3 grid(Nx / 64, Mx / 128);
        gemm_bf16_kernel<<<grid, 256, 0, stream>>>(A, W, bias, res, oF, oB, Mx, Nx, Kx, gelu);
    };

    // 0) downconvert weights + activations
    for (int i = 0; i < 10; ++i) cvt(wsrc[i], wb[i], wcnt[i]);
    cvt(x,   curb, (size_t)M * D);
    cvt(enc, encb, (size_t)M * D);

    // 1) masked self-attention
    gemm(curb, wb[0], m_bq, nullptr, nullptr, qb, M, D, D, 0);
    gemm(curb, wb[1], m_bk, nullptr, nullptr, kb, M, D, D, 0);
    gemm(curb, wb[2], m_bv, nullptr, nullptr, vb, M, D, D, 0);
    flash_attn_kernel<<<dim3(Nseq / 128, Bb * H), 256, 0, stream>>>(
        qb, kb, vb, attnb, Nseq, H, /*causal=*/1, scale);
    gemm(attnb, wb[3], m_bo, /*res=*/x, sumf, nullptr, M, D, D, 0);
    layernorm_kernel<<<M, 256, 0, stream>>>(sumf, ln1g, ln1b, curf, curb, D);

    // 2) cross-attention
    gemm(curb, wb[4], c_bq, nullptr, nullptr, qb, M, D, D, 0);
    gemm(encb, wb[5], c_bk, nullptr, nullptr, kb, M, D, D, 0);
    gemm(encb, wb[6], c_bv, nullptr, nullptr, vb, M, D, D, 0);
    flash_attn_kernel<<<dim3(Nseq / 128, Bb * H), 256, 0, stream>>>(
        qb, kb, vb, attnb, Nseq, H, /*causal=*/0, scale);
    gemm(attnb, wb[7], c_bo, /*res=*/curf, sumf, nullptr, M, D, D, 0);
    layernorm_kernel<<<M, 256, 0, stream>>>(sumf, ln2g, ln2b, curf, curb, D);

    // 3) feed-forward
    gemm(curb, wb[8], ffb1, nullptr, nullptr, hbb, M, F, D, /*gelu=*/1);
    gemm(hbb,  wb[9], ffb2, /*res=*/curf, sumf, nullptr, M, D, F, 0);
    layernorm_kernel<<<M, 256, 0, stream>>>(sumf, ln3g, ln3b, (float*)d_out, nullptr, D);
}
